// GauntTensorProductS2Grid_50036368999124
// MI455X (gfx1250) — matrix operand order repaired and checked by reading, not compiled, not run on hardware
//
#include <hip/hip_runtime.h>
#include <hip/hip_bf16.h>

// ---------------------------------------------------------------------------
// Gaunt tensor product via S2 grid, fused:
//   out[n,j] = sum_ba (X1[n,:]·Yin[:,ba]) * (X2[n,:]·Yin[:,ba]) * w[b] * Yout[j,ba]
// N=8192, dim_in=81 (pad 96), grid=64*128=8192, dim_out=289 (pad 304=19*16).
// bf16 WMMA (v_wmma_f32_16x16x32_bf16) with f32 accumulation, w folded into
// the staged Yout. Workspace requirement: ~6.6 MB.
// ---------------------------------------------------------------------------

typedef __attribute__((ext_vector_type(16))) __bf16 v16bf;
typedef __attribute__((ext_vector_type(8)))  float  v8f;
typedef __attribute__((ext_vector_type(2)))  float  f32x2;
typedef __attribute__((ext_vector_type(2)))  __bf16 bf16x2;

#define DIN   81
#define GRID  8192      // 64 beta * 128 alpha
#define DOUT  289
#define NJT   19        // 19 j-tiles of 16 (304)

// d_ws layout (bytes)
#define YIN_OFF   0u
#define YIN_SIZE  (512u * 3u * 1024u)          // 512 col16-chunks * 3 K-steps * 1KB frag
#define YOUT_OFF  YIN_SIZE
#define YOUT_SIZE (256u * 19u * 1024u)         // 256 col32-chunks * 19 j-tiles * 1KB frag

union V16 {
    unsigned int u[8];
    uint4        q[2];
    v16bf        v;
};

union BF2 {
    bf16x2       v;
    unsigned int u;
    __bf16       b[2];
};

// f32 pair -> packed bf16 pair (v_cvt_pk_bf16_f32)
__device__ __forceinline__ unsigned int pk_bf16(float x0, float x1) {
    f32x2 f; f[0] = x0; f[1] = x1;
    BF2 r; r.v = __builtin_convertvector(f, bf16x2);
    return r.u;
}

// ---------------------------------------------------------------------------
// Pre-pass 1: stage Yin[81,8192] (f32) as bf16 B-fragments (32x16 per frag).
// Fragment (c16, s): B[k][n] = Yin[32*s + k][c16*16 + n], k in [0,32), zero pad
// i>=81. Lane L stores its 16 bf16 values (k = 16*(L>>4) + 0..15, n = L&15)
// contiguously: frag is 32 lanes * 32B = 1KB.
// ---------------------------------------------------------------------------
__global__ void stage_yin_kernel(const float* __restrict__ yin,
                                 unsigned char* __restrict__ ws) {
    int gid  = blockIdx.x * blockDim.x + threadIdx.x;   // 512*3*32 threads
    int lane = gid & 31;
    int s    = (gid >> 5) % 3;
    int c16  = gid / (3 * 32);
    int col  = c16 * 16 + (lane & 15);
    int h    = lane >> 4;

    V16 t;
#pragma unroll
    for (int v = 0; v < 8; ++v) {
        int i0 = 32 * s + 16 * h + 2 * v;
        float x0 = (i0     < DIN) ? yin[(size_t)i0 * GRID + col]       : 0.0f;
        float x1 = (i0 + 1 < DIN) ? yin[(size_t)(i0 + 1) * GRID + col] : 0.0f;
        t.u[v] = pk_bf16(x0, x1);
    }

    uint4* dst = (uint4*)(ws + YIN_OFF + ((size_t)(c16 * 3 + s)) * 1024u + (size_t)lane * 32u);
    dst[0] = t.q[0];
    dst[1] = t.q[1];
}

// ---------------------------------------------------------------------------
// Pre-pass 2: stage w[b]*Yout[289,8192] as bf16 B-fragments (32x16 per frag).
// Fragment (c32, jt): B[k][j] = w[ba>>7]*Yout[jt*16+j][c32*32+k], zero pad
// j>=289. Same lane-major 1KB layout.
// ---------------------------------------------------------------------------
__global__ void stage_yout_kernel(const float* __restrict__ yout,
                                  const float* __restrict__ w,
                                  unsigned char* __restrict__ ws) {
    int gid  = blockIdx.x * blockDim.x + threadIdx.x;   // 256*19*32 threads
    int lane = gid & 31;
    int jt   = (gid >> 5) % NJT;
    int c32  = gid / (NJT * 32);
    int j    = jt * 16 + (lane & 15);
    int h    = lane >> 4;
    int ba0  = c32 * 32 + 16 * h;

    const float* row = yout + (size_t)j * GRID;
    V16 t;
#pragma unroll
    for (int v = 0; v < 8; ++v) {
        int ba = ba0 + 2 * v;
        float x0 = (j < DOUT) ? row[ba]     * w[ba >> 7]       : 0.0f;
        float x1 = (j < DOUT) ? row[ba + 1] * w[(ba + 1) >> 7] : 0.0f;
        t.u[v] = pk_bf16(x0, x1);
    }

    uint4* dst = (uint4*)(ws + YOUT_OFF + ((size_t)(c32 * NJT + jt)) * 1024u + (size_t)lane * 32u);
    dst[0] = t.q[0];
    dst[1] = t.q[1];
}

// ---------------------------------------------------------------------------
// Build A-fragment (16x32, bf16) for the coefficient matrices directly from
// global f32 rows. A layout: lane L holds row m=L&15; VGPR v -> k=16h+2v+{0,1}.
// Zero pad i>=81.
// ---------------------------------------------------------------------------
__device__ __forceinline__ v16bf build_a_frag(const float* __restrict__ row,
                                              int s, int h) {
    V16 t;
#pragma unroll
    for (int v = 0; v < 8; ++v) {
        int k0 = 32 * s + 16 * h + 2 * v;
        float x0 = (k0     < DIN) ? row[k0]     : 0.0f;
        float x1 = (k0 + 1 < DIN) ? row[k0 + 1] : 0.0f;
        t.u[v] = pk_bf16(x0, x1);
    }
    return t.v;
}

// ---------------------------------------------------------------------------
// Main fused kernel: 256 threads (8 waves), one 16-row N-tile per block.
// ---------------------------------------------------------------------------
__global__ void __launch_bounds__(256)
gaunt_s2grid_kernel(const float* __restrict__ x1,
                    const float* __restrict__ x2,
                    const unsigned char* __restrict__ ws,
                    float* __restrict__ out) {
    // double-buffered f tiles: [buf][chunk32][n(16)][k(32)] bf16 = 16 KB
    __shared__ __align__(16) __bf16 fbuf[2][8][16][32];

    const int tid  = threadIdx.x;
    const int wv   = __builtin_amdgcn_readfirstlane(tid >> 5);  // scalar wave id
    const int lane = tid & 31;
    const int col  = lane & 15;
    const int h    = lane >> 4;
    const int n0   = blockIdx.x * 16;  // N-row tile base

    const v16bf* yinf  = (const v16bf*)(ws + YIN_OFF);   // 32B fragments, lane-major
    const v16bf* youtf = (const v16bf*)(ws + YOUT_OFF);

    const v8f zero = {0.f, 0.f, 0.f, 0.f, 0.f, 0.f, 0.f, 0.f};

    // One-time: A-fragments of X1/X2 for this row tile (K = 96 -> 3 steps).
    v16bf a1[3], a2[3];
#pragma unroll
    for (int s = 0; s < 3; ++s) {
        a1[s] = build_a_frag(x1 + (size_t)(n0 + col) * DIN, s, h);
        a2[s] = build_a_frag(x2 + (size_t)(n0 + col) * DIN, s, h);
    }

    // j-tile ownership (scalar): waves 0..2 own 3 tiles, waves 3..7 own 2
    const int jstart = (wv < 3) ? 3 * wv : 9 + 2 * (wv - 3);
    const int jcnt   = (wv < 3) ? 3 : 2;
    v8f acc0 = zero, acc1 = zero, acc2 = zero;

    for (int it = 0; it < 32; ++it) {      // 32 iters * 256 grid cols = 8192
        const int p = it & 1;

        // ---- produce: this wave computes f for its 32-col chunk (2 tiles) ----
        // Force all 6 B-fragment loads to issue before the WMMA block so the
        // waits become partial (overlap later loads with earlier WMMAs).
        v16bf bfr[6];
#pragma unroll
        for (int t = 0; t < 2; ++t) {
            const int c16 = it * 16 + wv * 2 + t;
#pragma unroll
            for (int s = 0; s < 3; ++s)
                bfr[3 * t + s] = yinf[(size_t)(c16 * 3 + s) * 32 + lane];
        }
        __builtin_amdgcn_sched_barrier(0);

        v8f g1a = zero, g2a = zero, g1b = zero, g2b = zero;
#pragma unroll
        for (int s = 0; s < 3; ++s) {
            g1a = __builtin_amdgcn_wmma_f32_16x16x32_bf16(
                    false, a1[s], false, bfr[s],     (short)0, g1a, false, false);
            g2a = __builtin_amdgcn_wmma_f32_16x16x32_bf16(
                    false, a2[s], false, bfr[s],     (short)0, g2a, false, false);
            g1b = __builtin_amdgcn_wmma_f32_16x16x32_bf16(
                    false, a1[s], false, bfr[3 + s], (short)0, g1b, false, false);
            g2b = __builtin_amdgcn_wmma_f32_16x16x32_bf16(
                    false, a2[s], false, bfr[3 + s], (short)0, g2b, false, false);
        }

        // f = g1*g2, scatter transposed into LDS (C-layout -> [n][k])
        {
            const v8f fa = g1a * g2a;
            const v8f fb = g1b * g2b;
#pragma unroll
            for (int vp = 0; vp < 4; ++vp) {
                BF2 ca; ca.u = pk_bf16(fa[2 * vp], fa[2 * vp + 1]);
                BF2 cb; cb.u = pk_bf16(fb[2 * vp], fb[2 * vp + 1]);
                fbuf[p][wv][2 * vp     + 8 * h][col]      = ca.b[0];
                fbuf[p][wv][2 * vp + 1 + 8 * h][col]      = ca.b[1];
                fbuf[p][wv][2 * vp     + 8 * h][16 + col] = cb.b[0];
                fbuf[p][wv][2 * vp + 1 + 8 * h][16 + col] = cb.b[1];
            }
        }

        __syncthreads();

        // ---- consume: project all 8 chunks onto this wave's j-tiles ----
#pragma unroll
        for (int ch = 0; ch < 8; ++ch) {
            // A-fragment of f: two contiguous 16B LDS reads per lane
            V16 t;
            t.q[0] = *(const uint4*)(&fbuf[p][ch][col][16 * h]);
            t.q[1] = *(const uint4*)(&fbuf[p][ch][col][16 * h + 8]);
            const v16bf af = t.v;

            const size_t fb = (size_t)((it * 8 + ch) * NJT + jstart) * 32 + lane;
            // constant-trip loop + scalar guard -> static acc regs, full unroll
#pragma unroll
            for (int q = 0; q < 3; ++q) {
                if (q < jcnt) {
                    v16bf byo = youtf[fb + (size_t)q * 32];
                    v8f* accp = (q == 0) ? &acc0 : (q == 1) ? &acc1 : &acc2;
                    *accp = __builtin_amdgcn_wmma_f32_16x16x32_bf16(
                                false, af, false, byo, (short)0, *accp, false, false);
                }
            }
        }
    }

    // ---- store: C-layout -> out[n0+v+8h][16*jt+col], mask padded j ----
#pragma unroll
    for (int q = 0; q < 3; ++q) {
        if (q < jcnt) {
            const v8f acc = (q == 0) ? acc0 : (q == 1) ? acc1 : acc2;
            const int j = 16 * (jstart + q) + col;
            if (j < DOUT) {
#pragma unroll
                for (int v = 0; v < 8; ++v)
                    out[(size_t)(n0 + v + 8 * h) * DOUT + j] = acc[v];
            }
        }
    }
}

// ---------------------------------------------------------------------------
extern "C" void kernel_launch(void* const* d_in, const int* in_sizes, int n_in,
                              void* d_out, int out_size, void* d_ws, size_t ws_size,
                              hipStream_t stream) {
    const float* x1   = (const float*)d_in[0];   // [8192, 81]
    const float* x2   = (const float*)d_in[1];   // [8192, 81]
    const float* yin  = (const float*)d_in[2];   // [81, 64, 128]
    const float* yout = (const float*)d_in[3];   // [289, 64, 128]
    const float* w    = (const float*)d_in[4];   // [64]
    float* out = (float*)d_out;                  // [8192, 289]
    unsigned char* ws = (unsigned char*)d_ws;

    (void)in_sizes; (void)n_in; (void)out_size; (void)ws_size;

    // Stage Yin: 512*3*32 lanes = 49152 threads
    stage_yin_kernel<<<192, 256, 0, stream>>>(yin, ws);
    // Stage w*Yout: 256*19*32 lanes = 155648 threads
    stage_yout_kernel<<<608, 256, 0, stream>>>(yout, w, ws);
    // Fused GEMM-mul-GEMM: 512 row-tiles
    gaunt_s2grid_kernel<<<512, 256, 0, stream>>>(x1, x2, ws, out);
}
